// GCN_dis_25254407700904
// MI455X (gfx1250) — matrix operand order repaired and checked by reading, not compiled, not run on hardware
//
#include <hip/hip_runtime.h>
#include <hip/hip_bf16.h>

// ---------------------------------------------------------------------------
// Problem constants (from reference)
// ---------------------------------------------------------------------------
#define NNODE 6000      // ND == NR
#define FD    128       // feature dim
#define NH    4         // GAT heads
#define NE    144000    // edges
#define GATW  512       // H*F
#define NEG_SLOPE 0.2f

typedef __attribute__((ext_vector_type(16))) _Float16 v16h;
typedef __attribute__((ext_vector_type(8)))  _Float16 v8h;
typedef __attribute__((ext_vector_type(8)))  float    v8f;

// ---------------------------------------------------------------------------
// helpers
// ---------------------------------------------------------------------------
__device__ __forceinline__ unsigned fenc(float x) {
  unsigned u = __float_as_uint(x);
  return (u & 0x80000000u) ? ~u : (u | 0x80000000u);   // order-monotonic encoding
}
__device__ __forceinline__ float fdec(unsigned u) {
  return (u & 0x80000000u) ? __uint_as_float(u & 0x7fffffffu) : __uint_as_float(~u);
}
__device__ __forceinline__ float lrelu(float x) { return x > 0.f ? x : NEG_SLOPE * x; }

// ---------------------------------------------------------------------------
// Convert/transpose f32 -> f16 staging: dst[n*K + k] = src[off + n*rs + k*ks]
// ---------------------------------------------------------------------------
__global__ void k_cvt(const float* __restrict__ src, _Float16* __restrict__ dst,
                      int total, int K, int rs, int ks, int off) {
  int i = blockIdx.x * blockDim.x + threadIdx.x;
  if (i >= total) return;
  int n = i / K, k = i % K;
  dst[i] = (_Float16)src[off + (size_t)n * rs + (size_t)k * ks];
}

// ---------------------------------------------------------------------------
// WMMA GEMM: C[M,N] = A[M,K]*B[N,K]^T (+bias[n]) (+C)
// A,B f16 row-major, leading dim K (compile-time). One wave32 computes an
// (MT*16) x (NT*16) register-blocked tile: A/B fragments reused NT/MT times,
// K/32 fully unrolled -> MT*NT*K/32 static v_wmma_f32_16x16x32_f16.
// ---------------------------------------------------------------------------
template<int K, int MT, int NT, int LDC, bool BIAS, bool ACC>
__global__ __launch_bounds__(32)
void k_wmma_gemm(const _Float16* __restrict__ A, const _Float16* __restrict__ B,
                 const float* __restrict__ bias, float* __restrict__ C)
{
  const int lane = threadIdx.x & 31;
  const int g    = lane >> 4;     // half-wave group
  const int mr   = lane & 15;
  const int n0   = blockIdx.x * (16 * NT);
  const int m0   = blockIdx.y * (16 * MT);

  v8f acc[MT][NT];
#pragma unroll
  for (int mt = 0; mt < MT; ++mt)
#pragma unroll
    for (int nt = 0; nt < NT; ++nt) {
      if (ACC) {
#pragma unroll
        for (int i = 0; i < 8; ++i)
          acc[mt][nt][i] =
              C[(size_t)(m0 + mt * 16 + g * 8 + i) * LDC + (n0 + nt * 16 + mr)];
      } else {
#pragma unroll
        for (int i = 0; i < 8; ++i) acc[mt][nt][i] = 0.0f;
      }
    }

  const _Float16* Ap[MT];
  const _Float16* Bp[NT];
#pragma unroll
  for (int mt = 0; mt < MT; ++mt) Ap[mt] = A + (size_t)(m0 + mt * 16 + mr) * K;
#pragma unroll
  for (int nt = 0; nt < NT; ++nt) Bp[nt] = B + (size_t)(n0 + nt * 16 + mr) * K;

#pragma unroll
  for (int kb = 0; kb < K; kb += 32) {
    v16h a[MT], b[NT];
#pragma unroll
    for (int mt = 0; mt < MT; ++mt) {
      // ISA A 16x32 f16 layout: halves 0..7 -> K=g*8+j ; halves 8..15 -> K=16+g*8+j-8
      v8h lo = *(const v8h*)(Ap[mt] + kb + g * 8);
      v8h hi = *(const v8h*)(Ap[mt] + kb + 16 + g * 8);
#pragma unroll
      for (int j = 0; j < 8; ++j) { a[mt][j] = lo[j]; a[mt][j + 8] = hi[j]; }
    }
#pragma unroll
    for (int nt = 0; nt < NT; ++nt) {
      // ISA B 32x16 f16 layout: lane holds column n; halves 0..7 -> K=g*8+j ; halves 8..15 -> K=16+g*8+j-8
      v8h lo = *(const v8h*)(Bp[nt] + kb + g * 8);
      v8h hi = *(const v8h*)(Bp[nt] + kb + 16 + g * 8);
#pragma unroll
      for (int j = 0; j < 8; ++j) { b[nt][j] = lo[j]; b[nt][j + 8] = hi[j]; }
    }
#pragma unroll
    for (int mt = 0; mt < MT; ++mt)
#pragma unroll
      for (int nt = 0; nt < NT; ++nt)
        // 8 args: (neg_a, A, neg_b, B, c_mod, C, reuse_a, reuse_b)
        acc[mt][nt] = __builtin_amdgcn_wmma_f32_16x16x32_f16(
            false, a[mt], false, b[nt], (short)0, acc[mt][nt], false, false);
  }

#pragma unroll
  for (int nt = 0; nt < NT; ++nt) {
    float bv = BIAS ? bias[n0 + nt * 16 + mr] : 0.0f;
#pragma unroll
    for (int mt = 0; mt < MT; ++mt)
#pragma unroll
      for (int i = 0; i < 8; ++i)
        C[(size_t)(m0 + mt * 16 + g * 8 + i) * LDC + (n0 + nt * 16 + mr)] =
            acc[mt][nt][i] + bv;
  }
}

// ---------------------------------------------------------------------------
// elementwise / graph kernels
// ---------------------------------------------------------------------------
__global__ void k_fill(float* p, float v, int n) {
  int i = blockIdx.x * blockDim.x + threadIdx.x;
  if (i < n) p[i] = v;
}

__global__ void k_deg(const int* __restrict__ col, const float* __restrict__ ew,
                      float* __restrict__ deg, int n) {
  int i = blockIdx.x * blockDim.x + threadIdx.x;
  if (i < n) atomicAdd(&deg[col[i]], ew[i]);
}

__global__ void k_dinv(const float* __restrict__ deg, float* __restrict__ dinv, int n) {
  int i = blockIdx.x * blockDim.x + threadIdx.x;
  if (i < n) { float dg = deg[i]; dinv[i] = dg > 0.f ? rsqrtf(dg) : 0.f; }
}

__global__ void k_easum(const float* __restrict__ ew, float* __restrict__ scal, int n) {
  __shared__ float red[256];
  int i = blockIdx.x * blockDim.x + threadIdx.x;
  red[threadIdx.x] = (i < n) ? ew[i] : 0.f;
  __syncthreads();
  for (int s = 128; s > 0; s >>= 1) {
    if ((int)threadIdx.x < s) red[threadIdx.x] += red[threadIdx.x + s];
    __syncthreads();
  }
  if (threadIdx.x == 0) atomicAdd(&scal[0], red[0]);
}

// scal[1] = mean(ew); scal[2+h] = sum_f We[h*F+f]*a_edge[h*F+f]
__global__ void k_edot(const float* __restrict__ We, const float* __restrict__ ae,
                       float* __restrict__ scal) {
  __shared__ float red[FD];
  int t = threadIdx.x;
  for (int hh = 0; hh < NH; ++hh) {
    red[t] = We[hh * FD + t] * ae[hh * FD + t];
    __syncthreads();
    for (int s = FD / 2; s > 0; s >>= 1) {
      if (t < s) red[t] += red[t + s];
      __syncthreads();
    }
    if (t == 0) scal[2 + hh] = red[0];
    __syncthreads();
  }
  if (t == 0) scal[1] = scal[0] / (float)NE;
}

// GCN: out[n,f] = dinv[n]^2 * h[n,f] + b[f]   (self-loop term, full init)
__global__ void k_gcn_self(const float* __restrict__ h, const float* __restrict__ dinv,
                           const float* __restrict__ b, float* __restrict__ out, int n) {
  int i = blockIdx.x * blockDim.x + threadIdx.x;
  if (i >= n) return;
  int nd = i / FD, f = i % FD;
  float dv = dinv[nd];
  out[i] = dv * dv * h[i] + b[f];
}

// GCN edge scatter: out[c,f] += dinv[r]*ew*dinv[c] * h[r,f]
__global__ void k_gcn_scatter(const int* __restrict__ row, const int* __restrict__ col,
                              const float* __restrict__ ew, const float* __restrict__ dinv,
                              const float* __restrict__ h, float* __restrict__ out, int n) {
  int i = blockIdx.x * blockDim.x + threadIdx.x;
  if (i >= n) return;
  int e = i / FD, f = i % FD;
  int r = row[e], c = col[e];
  float coef = dinv[r] * ew[e] * dinv[c];
  atomicAdd(&out[(size_t)c * FD + f], coef * h[(size_t)r * FD + f]);
}

// GAT per-node attention dots: s[n,h], d[n,h]
__global__ void k_gat_sd(const float* __restrict__ hgat, const float* __restrict__ asrc,
                         const float* __restrict__ adst, float* __restrict__ s,
                         float* __restrict__ d, int n) {
  int i = blockIdx.x * blockDim.x + threadIdx.x;
  if (i >= n) return;
  int nd = i / NH, hh = i % NH;
  const float* hp = hgat + (size_t)nd * GATW + hh * FD;
  const float* ap = asrc + hh * FD;
  const float* bp = adst + hh * FD;
  float ss = 0.f, dd = 0.f;
  for (int f = 0; f < FD; ++f) { float hv = hp[f]; ss += hv * ap[f]; dd += hv * bp[f]; }
  s[i] = ss; d[i] = dd;
}

// per-edge per-head logits (leaky-relu'd)
__global__ void k_alpha(const int* __restrict__ row, const int* __restrict__ col,
                        const float* __restrict__ ew, const float* __restrict__ s,
                        const float* __restrict__ d, const float* __restrict__ scal,
                        float* __restrict__ alpha, int n) {
  int i = blockIdx.x * blockDim.x + threadIdx.x;
  if (i >= n) return;
  int e = i / NH, hh = i % NH;
  alpha[i] = lrelu(s[row[e] * NH + hh] + d[col[e] * NH + hh] + ew[e] * scal[2 + hh]);
}

// init per-(node,head) max with self-loop alpha; also stash selfa
__global__ void k_minit(const float* __restrict__ s, const float* __restrict__ d,
                        const float* __restrict__ scal, float* __restrict__ selfa,
                        unsigned* __restrict__ mE, int n) {
  int i = blockIdx.x * blockDim.x + threadIdx.x;
  if (i >= n) return;
  int hh = i % NH;
  float a = lrelu(s[i] + d[i] + scal[1] * scal[2 + hh]);
  selfa[i] = a;
  mE[i] = fenc(a);
}

__global__ void k_mmax(const int* __restrict__ col, const float* __restrict__ alpha,
                       unsigned* __restrict__ mE, int n) {
  int i = blockIdx.x * blockDim.x + threadIdx.x;
  if (i >= n) return;
  int e = i / NH, hh = i % NH;
  atomicMax(&mE[col[e] * NH + hh], fenc(alpha[i]));
}

__global__ void k_esuminit(const float* __restrict__ selfa, const unsigned* __restrict__ mE,
                           float* __restrict__ esum, int n) {
  int i = blockIdx.x * blockDim.x + threadIdx.x;
  if (i < n) esum[i] = expf(selfa[i] - fdec(mE[i]));
}

__global__ void k_esumadd(const int* __restrict__ col, const float* __restrict__ alpha,
                          const unsigned* __restrict__ mE, float* __restrict__ esum, int n) {
  int i = blockIdx.x * blockDim.x + threadIdx.x;
  if (i >= n) return;
  int e = i / NH, hh = i % NH;
  int t = col[e] * NH + hh;
  atomicAdd(&esum[t], expf(alpha[i] - fdec(mE[t])));
}

__global__ void k_coef(const int* __restrict__ col, const float* __restrict__ alpha,
                       const unsigned* __restrict__ mE, const float* __restrict__ esum,
                       float* __restrict__ coef, int n) {
  int i = blockIdx.x * blockDim.x + threadIdx.x;
  if (i >= n) return;
  int e = i / NH, hh = i % NH;
  int t = col[e] * NH + hh;
  coef[i] = expf(alpha[i] - fdec(mE[t])) / fmaxf(esum[t], 1e-16f);
}

// GAT self term + bias (full init): out[n,f] = b[f] + (1/H) sum_h selfcoef*hgat[n,h,f]
__global__ void k_gat_self(const float* __restrict__ hgat, const float* __restrict__ selfa,
                           const unsigned* __restrict__ mE, const float* __restrict__ esum,
                           const float* __restrict__ b, float* __restrict__ out, int n) {
  int i = blockIdx.x * blockDim.x + threadIdx.x;
  if (i >= n) return;
  int nd = i / FD, f = i % FD;
  float acc = b[f];
#pragma unroll
  for (int hh = 0; hh < NH; ++hh) {
    int t = nd * NH + hh;
    float cc = expf(selfa[t] - fdec(mE[t])) / fmaxf(esum[t], 1e-16f);
    acc += 0.25f * cc * hgat[(size_t)nd * GATW + hh * FD + f];
  }
  out[i] = acc;
}

// GAT edge scatter: out[c,f] += (1/H) sum_h coef[e,h]*hgat[r,h,f]
__global__ void k_gat_scatter(const int* __restrict__ row, const int* __restrict__ col,
                              const float* __restrict__ coef, const float* __restrict__ hgat,
                              float* __restrict__ out, int n) {
  int i = blockIdx.x * blockDim.x + threadIdx.x;
  if (i >= n) return;
  int e = i / FD, f = i % FD;
  int r = row[e], c = col[e];
  float v = 0.f;
#pragma unroll
  for (int hh = 0; hh < NH; ++hh)
    v += coef[e * NH + hh] * hgat[(size_t)r * GATW + hh * FD + f];
  atomicAdd(&out[(size_t)c * FD + f], 0.25f * v);
}

// x = relu(0.5*(a+b))
__global__ void k_combine(const float* __restrict__ a, const float* __restrict__ b,
                          float* __restrict__ x, int n) {
  int i = blockIdx.x * blockDim.x + threadIdx.x;
  if (i < n) x[i] = fmaxf(0.5f * (a[i] + b[i]), 0.f);
}

// ---------------------------------------------------------------------------
// host side
// ---------------------------------------------------------------------------
static inline unsigned nblk(long n, int tb) { return (unsigned)((n + tb - 1) / tb); }

struct Ws {
  float *h, *hgat, *s, *d, *deg, *dinv, *selfa, *esum, *alpha, *coef;
  float *gcno, *gato, *x1, *x2, *scal;
  unsigned* mE;
  _Float16 *ha, *hb;   // f16 staging for GEMM A / B operands
};

static Ws make_ws(void* d_ws) {
  float* W = (float*)d_ws;
  Ws w;
  w.h     = W;                 // 768000
  w.hgat  = W + 768000;        // 3072000
  w.s     = W + 3840000;       // 24000
  w.d     = W + 3864000;       // 24000
  w.deg   = W + 3888000;       // 6000
  w.dinv  = W + 3894000;       // 6000
  w.selfa = W + 3900000;       // 24000
  w.mE    = (unsigned*)(W + 3924000); // 24000
  w.esum  = W + 3948000;       // 24000
  w.alpha = W + 3972000;       // 576000
  w.coef  = W + 4548000;       // 576000
  w.gcno  = W + 5124000;       // 768000
  w.gato  = W + 5892000;       // 768000
  w.x1    = W + 6660000;       // 768000
  w.x2    = W + 7428000;       // 768000
  w.scal  = W + 8196000;       // 8
  w.ha    = (_Float16*)(W + 8200000);  // 768000 halves (16B aligned)
  w.hb    = w.ha + 768000;             // 768000 halves
  return w;
}

static void run_branch(const float* x, const int* ei, const float* ew,
                       void* const* prm, float* fea, const Ws& w, hipStream_t st)
{
  const float* g1W = (const float*)prm[0];
  const float* g1b = (const float*)prm[1];
  const float* g2W = (const float*)prm[2];
  const float* g2b = (const float*)prm[3];
  const float* gW  = (const float*)prm[4];
  const float* gas = (const float*)prm[5];
  const float* gad = (const float*)prm[6];
  const float* gWe = (const float*)prm[7];
  const float* gae = (const float*)prm[8];
  const float* gb  = (const float*)prm[9];
  const float* cW  = (const float*)prm[10];
  const float* cb  = (const float*)prm[11];

  const int* row = ei;
  const int* col = ei + NE;
  const int TB = 256;

  // symmetric-norm degrees (shared by both GCN layers)
  k_fill<<<nblk(NNODE, TB), TB, 0, st>>>(w.deg, 1.0f, NNODE);   // self-loop weight 1
  k_deg<<<nblk(NE, TB), TB, 0, st>>>(col, ew, w.deg, NE);
  k_dinv<<<nblk(NNODE, TB), TB, 0, st>>>(w.deg, w.dinv, NNODE);

  // edge-attr scalars: mean(ew) and edot[h]
  k_fill<<<1, 32, 0, st>>>(w.scal, 0.0f, 8);
  k_easum<<<nblk(NE, TB), TB, 0, st>>>(ew, w.scal, NE);
  k_edot<<<1, FD, 0, st>>>(gWe, gae, w.scal);

  for (int layer = 0; layer < 2; ++layer) {
    const float* xin = (layer == 0) ? x : w.x1;
    const float* gcw = (layer == 0) ? g1W : g2W;
    const float* gcb = (layer == 0) ? g1b : g2b;
    float* xout = (layer == 0) ? w.x1 : w.x2;

    // stage activations once per layer (used by GCN and GAT GEMMs)
    k_cvt<<<nblk((long)NNODE * FD, TB), TB, 0, st>>>(xin, w.ha, NNODE * FD, FD, FD, 1, 0);

    // ---- GCN ----  B^T: hb[n*K+k] = Wg[k*N+n]
    k_cvt<<<nblk(FD * FD, TB), TB, 0, st>>>(gcw, w.hb, FD * FD, FD, 1, FD, 0);
    k_wmma_gemm<FD, 1, 1, FD, false, false><<<dim3(FD / 16, NNODE / 16), 32, 0, st>>>(
        w.ha, w.hb, nullptr, w.h);
    k_gcn_self<<<nblk((long)NNODE * FD, TB), TB, 0, st>>>(w.h, w.dinv, gcb, w.gcno,
                                                          NNODE * FD);
    k_gcn_scatter<<<nblk((long)NE * FD, TB), TB, 0, st>>>(row, col, ew, w.dinv, w.h,
                                                          w.gcno, NE * FD);

    // ---- GAT ----
    k_cvt<<<nblk(GATW * FD, TB), TB, 0, st>>>(gW, w.hb, GATW * FD, FD, 1, GATW, 0);
    k_wmma_gemm<FD, 1, 1, GATW, false, false><<<dim3(GATW / 16, NNODE / 16), 32, 0, st>>>(
        w.ha, w.hb, nullptr, w.hgat);
    k_gat_sd<<<nblk((long)NNODE * NH, TB), TB, 0, st>>>(w.hgat, gas, gad, w.s, w.d,
                                                        NNODE * NH);
    k_alpha<<<nblk((long)NE * NH, TB), TB, 0, st>>>(row, col, ew, w.s, w.d, w.scal,
                                                    w.alpha, NE * NH);
    k_minit<<<nblk((long)NNODE * NH, TB), TB, 0, st>>>(w.s, w.d, w.scal, w.selfa, w.mE,
                                                       NNODE * NH);
    k_mmax<<<nblk((long)NE * NH, TB), TB, 0, st>>>(col, w.alpha, w.mE, NE * NH);
    k_esuminit<<<nblk((long)NNODE * NH, TB), TB, 0, st>>>(w.selfa, w.mE, w.esum,
                                                          NNODE * NH);
    k_esumadd<<<nblk((long)NE * NH, TB), TB, 0, st>>>(col, w.alpha, w.mE, w.esum,
                                                      NE * NH);
    k_coef<<<nblk((long)NE * NH, TB), TB, 0, st>>>(col, w.alpha, w.mE, w.esum, w.coef,
                                                   NE * NH);
    k_gat_self<<<nblk((long)NNODE * FD, TB), TB, 0, st>>>(w.hgat, w.selfa, w.mE, w.esum,
                                                          gb, w.gato, NNODE * FD);
    k_gat_scatter<<<nblk((long)NE * FD, TB), TB, 0, st>>>(row, col, w.coef, w.hgat,
                                                          w.gato, NE * FD);

    // ---- combine ----
    k_combine<<<nblk((long)NNODE * FD, TB), TB, 0, st>>>(w.gcno, w.gato, xout, NNODE * FD);
  }

  // CNN head: fea = x1 @ cW[:,0,:,0]^T + x2 @ cW[:,1,:,0]^T + cb
  // cW element (o,l,f) at o*(2F) + l*F + f  ->  hb[o*K+k] with rs=2F, ks=1, off=l*F
  k_cvt<<<nblk((long)NNODE * FD, TB), TB, 0, st>>>(w.x1, w.ha, NNODE * FD, FD, FD, 1, 0);
  k_cvt<<<nblk(FD * FD, TB), TB, 0, st>>>(cW, w.hb, FD * FD, FD, 2 * FD, 1, 0);
  k_wmma_gemm<FD, 1, 1, FD, true, false><<<dim3(FD / 16, NNODE / 16), 32, 0, st>>>(
      w.ha, w.hb, cb, fea);
  k_cvt<<<nblk((long)NNODE * FD, TB), TB, 0, st>>>(w.x2, w.ha, NNODE * FD, FD, FD, 1, 0);
  k_cvt<<<nblk(FD * FD, TB), TB, 0, st>>>(cW, w.hb, FD * FD, FD, 2 * FD, 1, FD);
  k_wmma_gemm<FD, 1, 1, FD, false, true><<<dim3(FD / 16, NNODE / 16), 32, 0, st>>>(
      w.ha, w.hb, nullptr, fea);
}

extern "C" void kernel_launch(void* const* d_in, const int* in_sizes, int n_in,
                              void* d_out, int out_size, void* d_ws, size_t ws_size,
                              hipStream_t stream)
{
  (void)in_sizes; (void)n_in; (void)out_size; (void)ws_size;

  const float* x_dis  = (const float*)d_in[0];
  const float* x_drug = (const float*)d_in[1];
  const int*   ei_dis  = (const int*)d_in[2];
  const int*   ei_drug = (const int*)d_in[3];
  const float* ew_dis  = (const float*)d_in[4];
  const float* ew_drug = (const float*)d_in[5];

  float* out      = (float*)d_out;
  float* fea_dis  = out + (size_t)NNODE * NNODE;          // [6000,128]
  float* fea_drug = fea_dis + (size_t)NNODE * FD;         // [6000,128]

  Ws w = make_ws(d_ws);
  const int TB = 256;

  run_branch(x_dis,  ei_dis,  ew_dis,  d_in + 6,  fea_dis,  w, stream);
  run_branch(x_drug, ei_drug, ew_drug, d_in + 18, fea_drug, w, stream);

  // scores = drug_fea @ dis_fea^T : B[N,K] view is exactly dis_fea[6000,128]
  // 48x48 register-blocked wave tile: 3x A-frag reuse, 3x B-frag reuse,
  // 36 static WMMAs, 3x less L2 fragment traffic on the dominant GEMM.
  k_cvt<<<nblk((long)NNODE * FD, TB), TB, 0, stream>>>(fea_drug, w.ha, NNODE * FD, FD,
                                                       FD, 1, 0);
  k_cvt<<<nblk((long)NNODE * FD, TB), TB, 0, stream>>>(fea_dis, w.hb, NNODE * FD, FD,
                                                       FD, 1, 0);
  k_wmma_gemm<FD, 3, 3, NNODE, false, false>
      <<<dim3(NNODE / 48, NNODE / 48), 32, 0, stream>>>(w.ha, w.hb, nullptr, out);
}
